// NLinear_87351044866622
// MI455X (gfx1250) — matrix operand order repaired and checked by reading, not compiled, not run on hardware
//
#include <hip/hip_runtime.h>
#include <hip/hip_bf16.h>
#include <math.h>

// ---- problem constants (match reference) ----
#define N_STATIONS 276
#define CTX        384
#define TGT        96
#define BATCH      4096
#define TILE_M     16
#define MAX_TILES  (BATCH / TILE_M + N_STATIONS)   // 532 worst case

typedef __attribute__((ext_vector_type(16))) __bf16 v16bf;
typedef __attribute__((ext_vector_type(8)))  float  v8f;

// ---- workspace layout (all int32) ----
struct Ws {
    int cnt[N_STATIONS];
    int cursor[N_STATIONS];
    int rowOff[N_STATIONS];
    int tileStation[MAX_TILES];
    int tileRowStart[MAX_TILES];
    int tileRowCnt[MAX_TILES];
    int totalTiles;
    int sorted[BATCH];
};

// Pass 0: re-zero counters every call (ws is poisoned once, never re-poisoned)
__global__ void nl_zero(Ws* __restrict__ ws) {
    int i = blockIdx.x * 256 + threadIdx.x;
    if (i < N_STATIONS) { ws->cnt[i] = 0; ws->cursor[i] = 0; }
    if (i == 0) ws->totalTiles = 0;
}

// Pass 1: histogram samples per station
__global__ void nl_count(const int* __restrict__ stations, Ws* __restrict__ ws) {
    int i = blockIdx.x * 256 + threadIdx.x;
    if (i < BATCH) atomicAdd(&ws->cnt[stations[i]], 1);
}

// Pass 2: serial scan over 276 stations; build row offsets and tile table
__global__ void nl_scan(Ws* __restrict__ ws) {
    if (threadIdx.x == 0 && blockIdx.x == 0) {
        int ro = 0, to = 0;
        for (int s = 0; s < N_STATIONS; ++s) {
            int c = ws->cnt[s];
            ws->rowOff[s] = ro;
            int nt = (c + TILE_M - 1) >> 4;
            for (int j = 0; j < nt; ++j) {
                ws->tileStation[to + j]  = s;
                ws->tileRowStart[to + j] = ro + j * TILE_M;
                int rem = c - j * TILE_M;
                ws->tileRowCnt[to + j]   = rem < TILE_M ? rem : TILE_M;
            }
            ro += c;
            to += nt;
        }
        ws->totalTiles = to;
    }
}

// Pass 3: scatter sample ids into station-sorted order
__global__ void nl_scatter(const int* __restrict__ stations, Ws* __restrict__ ws) {
    int i = blockIdx.x * 256 + threadIdx.x;
    if (i < BATCH) {
        int s = stations[i];
        int p = atomicAdd(&ws->cursor[s], 1);
        ws->sorted[ws->rowOff[s] + p] = i;
    }
}

// Pass 4: per-tile GEMM.  Block = 6 waves (192 threads); wave w owns output
// columns [16w, 16w+16).  K=384 swept with 12x v_wmma_f32_16x16x32_bf16.
__global__ __launch_bounds__(192) void nl_gemm(
    const float* __restrict__ z, const float* __restrict__ W,
    const float* __restrict__ bias, const float* __restrict__ loc,
    const float* __restrict__ scale, float* __restrict__ out,
    const Ws* __restrict__ ws)
{
    int t = blockIdx.x;
    if (t >= ws->totalTiles) return;           // uniform per block

    const int s        = ws->tileStation[t];
    const int rowStart = ws->tileRowStart[t];
    const int rowCnt   = ws->tileRowCnt[t];

    // A tile staged in the exact 16-bit A-matrix (16x32) lane layout:
    // ldsA[kblock][lane][halfIdx], 12 KB total.
    __shared__ __align__(32) __bf16 ldsA[CTX / 32][32][16];
    __shared__ int smpl[TILE_M];

    const float locS = loc[s];
    const float sclS = scale[s];
    const float invS = 1.0f / sclS;

    const int tid = threadIdx.x;
    if (tid < TILE_M)
        smpl[tid] = (tid < rowCnt) ? ws->sorted[rowStart + tid] : -1;
    __syncthreads();

    // Stage normalized activations: zn = (z - loc)/scale, converted to bf16,
    // swizzled into A-fragment order (lane<16: K%32 in {0..7,16..23};
    // lane>=16: K%32 in {8..15,24..31}).
    for (int e = tid; e < TILE_M * CTX; e += 192) {
        int m = e / CTX;
        int k = e - m * CTX;
        int si = smpl[m];
        float v = 0.0f;
        if (si >= 0) v = (z[(size_t)si * CTX + k] - locS) * invS;
        int kk   = k & 31;
        int kb   = k >> 5;
        int lane = m + ((kk & 8) ? 16 : 0);
        int idx  = (kk & 7) + ((kk & 16) >> 1);      // 0..7 or 8..15
        ldsA[kb][lane][idx] = (__bf16)v;
    }
    __syncthreads();

    const int wave = tid >> 5;          // 0..5 -> N-subtile
    const int lane = tid & 31;
    const int ncol = lane & 15;
    const int n    = wave * 16 + ncol;  // global output column (0..95)

    // B fragment source: row n of W[s] is column n of the 32x16 B matrix.
    // lanes 0-15 carry K {0..7,16..23}, lanes 16-31 carry K {8..15,24..31}.
    const float* wrow = W + ((size_t)s * TGT + n) * CTX + ((lane & 16) ? 8 : 0);

    v8f acc = {0.f, 0.f, 0.f, 0.f, 0.f, 0.f, 0.f, 0.f};

#pragma unroll
    for (int kb = 0; kb < CTX / 32; ++kb) {
        v16bf a = *(const v16bf*)&ldsA[kb][lane][0];

        const float4* wp = (const float4*)(wrow + kb * 32);
        float4 w0 = wp[0], w1 = wp[1], w2 = wp[4], w3 = wp[5];
        v16bf bf;
        bf[0]  = (__bf16)w0.x; bf[1]  = (__bf16)w0.y;
        bf[2]  = (__bf16)w0.z; bf[3]  = (__bf16)w0.w;
        bf[4]  = (__bf16)w1.x; bf[5]  = (__bf16)w1.y;
        bf[6]  = (__bf16)w1.z; bf[7]  = (__bf16)w1.w;
        bf[8]  = (__bf16)w2.x; bf[9]  = (__bf16)w2.y;
        bf[10] = (__bf16)w2.z; bf[11] = (__bf16)w2.w;
        bf[12] = (__bf16)w3.x; bf[13] = (__bf16)w3.y;
        bf[14] = (__bf16)w3.z; bf[15] = (__bf16)w3.w;

        acc = __builtin_amdgcn_wmma_f32_16x16x32_bf16(
            /*neg_a=*/false, a, /*neg_b=*/false, bf,
            /*c_mod=*/(short)0, acc, /*reuse_a=*/false, /*reuse_b=*/false);
    }

    // Epilogue: C/D layout -> VGPR r holds M = r + 8*(lane>=16), N = lane&15.
    const float bn    = bias[(size_t)s * TGT + n];
    const int   mbase = (lane & 16) ? 8 : 0;
#pragma unroll
    for (int r = 0; r < 8; ++r) {
        int m  = mbase + r;
        int si = smpl[m];
        if (si >= 0) {
            float v  = (acc[r] + bn) * sclS + locS;
            float sp = (v > 20.0f) ? v : log1pf(__expf(v));   // softplus
            out[(size_t)si * TGT + n] = sp;
        }
    }
}

extern "C" void kernel_launch(void* const* d_in, const int* in_sizes, int n_in,
                              void* d_out, int out_size, void* d_ws, size_t ws_size,
                              hipStream_t stream) {
    const float* z        = (const float*)d_in[0];
    const int*   stations = (const int*)d_in[1];   // harness passes integers as int32
    const float* W        = (const float*)d_in[2];
    const float* b        = (const float*)d_in[3];
    const float* loc      = (const float*)d_in[4];
    const float* scale    = (const float*)d_in[5];
    float*       out      = (float*)d_out;
    Ws*          ws       = (Ws*)d_ws;

    nl_zero   <<<(N_STATIONS + 255) / 256, 256, 0, stream>>>(ws);
    nl_count  <<<(BATCH + 255) / 256,      256, 0, stream>>>(stations, ws);
    nl_scan   <<<1, 32, 0, stream>>>(ws);
    nl_scatter<<<(BATCH + 255) / 256,      256, 0, stream>>>(stations, ws);
    nl_gemm   <<<MAX_TILES, 192, 0, stream>>>(z, W, b, loc, scale, out, ws);
}
